// FlaxSelectiveAttention_18305150616342
// MI455X (gfx1250) — matrix operand order repaired and checked by reading, not compiled, not run on hardware
//
#include <hip/hip_runtime.h>
#include <hip/hip_bf16.h>

typedef __attribute__((ext_vector_type(16))) _Float16 v16h;
typedef __attribute__((ext_vector_type(8)))  float    v8f;
typedef __attribute__((ext_vector_type(8)))  _Float16 h8;
typedef __attribute__((ext_vector_type(4)))  _Float16 h4;
typedef __attribute__((ext_vector_type(2)))  _Float16 h2;
typedef __attribute__((ext_vector_type(4)))  float    f4;
typedef __attribute__((ext_vector_type(2)))  float    f2;

// Problem constants
constexpr int B_  = 2;
constexpr int S_  = 4096;
constexpr int H_  = 512;
constexpr int G_  = 18;
constexpr int NH_ = 8;
constexpr int D_  = 64;
constexpr int M_  = B_ * S_;        // 8192 tokens
constexpr int GK_ = G_ * H_;        // 9216 (contraction dim & KV output dim)

// GEMM tiling: block = 128 threads (4 waves), block tile 64(M) x 128(N),
// per-wave tile 32x64 = 2x4 WMMA fragments, k-step 32.
constexpr int TM  = 64;
constexpr int TN  = 128;
constexpr int TK  = 32;
constexpr int LDP = TK + 8;         // padded LDS row stride in halfs (80 B, 16B-aligned)

// ---------------------------------------------------------------------------
// Kernel 1: gather connected tokens -> A (f16), zero rows where conn == -1
// ---------------------------------------------------------------------------
__global__ __launch_bounds__(128)
void gather_kernel(const float* __restrict__ hidden,
                   const int* __restrict__ conn,
                   _Float16* __restrict__ A) {
    int idx = blockIdx.x;                 // b*S*G + s*G + g
    int g   = idx % G_;
    int bs  = idx / G_;                   // b*S + s
    int c   = conn[idx];
    _Float16* dst = A + (size_t)bs * GK_ + (size_t)g * H_;
    int i = threadIdx.x;                  // 128 threads x 4 floats = 512 = H_
    h4 hv;
    if (c != -1) {
        int cc = c < 0 ? 0 : (c >= S_ ? S_ - 1 : c);
        int b  = bs / S_;
        const float* src = hidden + ((size_t)b * S_ + cc) * H_;
        f4 w = *(const f4*)&src[i * 4];
        hv[0] = (_Float16)w[0]; hv[1] = (_Float16)w[1];
        hv[2] = (_Float16)w[2]; hv[3] = (_Float16)w[3];
    } else {
        hv[0] = hv[1] = hv[2] = hv[3] = (_Float16)0.0f;
    }
    *(h4*)&dst[i * 4] = hv;
}

// --- weight-tile staging: global f32 -> registers -> (f16, transposed) LDS --
__device__ __forceinline__
void load_w_tile(const float* __restrict__ Bw, int N, int k0, int n0, int tid,
                 f4 (&r)[2][4]) {
    const int nb = (tid & 31) * 4;
    const int kb = (tid >> 5) * 4;
    #pragma unroll
    for (int p = 0; p < 2; ++p) {
        const int kk = kb + p * 16;
        const float* gp = &Bw[(size_t)(k0 + kk) * N + (n0 + nb)];
        r[p][0] = *(const f4*)(gp);
        r[p][1] = *(const f4*)(gp + N);
        r[p][2] = *(const f4*)(gp + 2 * (size_t)N);
        r[p][3] = *(const f4*)(gp + 3 * (size_t)N);
    }
}

__device__ __forceinline__
void store_w_tile(_Float16* __restrict__ Bs, int tid, const f4 (&r)[2][4]) {
    const int nb = (tid & 31) * 4;
    const int kb = (tid >> 5) * 4;
    #pragma unroll
    for (int p = 0; p < 2; ++p) {
        const int kk = kb + p * 16;
        #pragma unroll
        for (int i = 0; i < 4; ++i) {
            h4 cvec;
            cvec[0] = (_Float16)r[p][0][i]; cvec[1] = (_Float16)r[p][1][i];
            cvec[2] = (_Float16)r[p][2][i]; cvec[3] = (_Float16)r[p][3][i];
            *(h4*)&Bs[(nb + i) * LDP + kk] = cvec;
        }
    }
}

// ---------------------------------------------------------------------------
// Kernel 2/5: C[M,N] (f32) = A[M,K] (f32) @ B[K,N] (f32), WMMA f16 path.
// Software-pipelined: next tile's global loads issue before the WMMA phase.
// ---------------------------------------------------------------------------
__global__ __launch_bounds__(128)
void gemm_f32_wmma(const float* __restrict__ A, const float* __restrict__ Bw,
                   float* __restrict__ C, int M, int N, int K) {
    __shared__ _Float16 As[TM * LDP];
    __shared__ _Float16 Bs[TN * LDP];     // transposed: Bs[n*LDP + k]
    const int m0   = blockIdx.x * TM;
    const int n0   = blockIdx.y * TN;
    const int tid  = threadIdx.x;
    const int lane = tid & 31;
    const int wave = tid >> 5;
    const int wm   = (wave >> 1) * 32;    // wave sub-tile origin (M)
    const int wn   = (wave & 1) * 64;     // wave sub-tile origin (N)
    const int mrow = lane & 15;
    const int half = lane >> 4;
    (void)M;

    const int ar = tid >> 1, ac = (tid & 1) * 16;   // A chunk: 2 per row pair
    v8f acc[2][4] = {};
    f4 aReg[4];                // 64x32 A tile: 128 thr * 4 f4 = 16 f32 each
    f4 wReg[2][4];

    // prologue: load tile 0
    #pragma unroll
    for (int t = 0; t < 4; ++t)
        aReg[t] = *(const f4*)&A[(size_t)(m0 + ar) * K + (0 + ac + t * 4)];
    load_w_tile(Bw, N, 0, n0, tid, wReg);

    for (int k0 = 0; k0 < K; k0 += TK) {
        // store staged registers into LDS (f32 -> f16 conversion here)
        #pragma unroll
        for (int t = 0; t < 4; ++t) {
            h4 hv;
            hv[0] = (_Float16)aReg[t][0]; hv[1] = (_Float16)aReg[t][1];
            hv[2] = (_Float16)aReg[t][2]; hv[3] = (_Float16)aReg[t][3];
            *(h4*)&As[ar * LDP + ac + t * 4] = hv;
        }
        store_w_tile(Bs, tid, wReg);
        __syncthreads();

        // prefetch next tile (loads overlap the WMMA phase below)
        if (k0 + TK < K) {
            #pragma unroll
            for (int t = 0; t < 4; ++t)
                aReg[t] = *(const f4*)&A[(size_t)(m0 + ar) * K + (k0 + TK + ac + t * 4)];
            load_w_tile(Bw, N, k0 + TK, n0, tid, wReg);
        }

        v16h af[2], bf[4];
        #pragma unroll
        for (int i = 0; i < 2; ++i) {
            const _Float16* ap = &As[(wm + i * 16 + mrow) * LDP + half * 16];
            #pragma unroll
            for (int e = 0; e < 8; ++e) { af[i][e] = ap[e]; af[i][e + 8] = ap[8 + e]; }
        }
        #pragma unroll
        for (int j = 0; j < 4; ++j) {
            const _Float16* bp = &Bs[(wn + j * 16 + mrow) * LDP + half * 16];
            #pragma unroll
            for (int e = 0; e < 16; ++e) bf[j][e] = bp[e];
        }
        #pragma unroll
        for (int i = 0; i < 2; ++i)
            #pragma unroll
            for (int j = 0; j < 4; ++j)
                acc[i][j] = __builtin_amdgcn_wmma_f32_16x16x32_f16(
                    false, af[i], false, bf[j], (short)0, acc[i][j], false, false);
        __syncthreads();
    }

    const int ncol = lane & 15, rbase = (lane >> 4) * 8;
    #pragma unroll
    for (int i = 0; i < 2; ++i)
        #pragma unroll
        for (int j = 0; j < 4; ++j)
            #pragma unroll
            for (int v = 0; v < 8; ++v) {
                int row = m0 + wm + i * 16 + rbase + v;
                int col = n0 + wn + j * 16 + ncol;
                C[(size_t)row * N + col] = acc[i][j][v];
            }
}

// ---------------------------------------------------------------------------
// Kernel 3: dual GEMM  K = A@Wk, V = A@Wv  (A f16 ws, W f32 global, out f16)
// M = 8192, N = K = 9216.  Grid: (M/64 fast, N/128) so weight N-panels stay
// in the 192MB L2; the 151MB f16 A matrix is L2-resident across N panels.
// ---------------------------------------------------------------------------
__global__ __launch_bounds__(128)
void gemm_kv_wmma(const _Float16* __restrict__ A,
                  const float* __restrict__ Wk, const float* __restrict__ Wv,
                  _Float16* __restrict__ Ko, _Float16* __restrict__ Vo) {
    __shared__ _Float16 As [TM * LDP];
    __shared__ _Float16 Bks[TN * LDP];    // transposed
    __shared__ _Float16 Bvs[TN * LDP];    // transposed
    const int N = GK_, K = GK_;
    const int m0   = blockIdx.x * TM;
    const int n0   = blockIdx.y * TN;
    const int tid  = threadIdx.x;
    const int lane = tid & 31;
    const int wave = tid >> 5;
    const int wm   = (wave >> 1) * 32;
    const int wn   = (wave & 1) * 64;
    const int mrow = lane & 15;
    const int half = lane >> 4;

    const int ar = tid >> 1, ac = (tid & 1) * 16;   // A: 16 halfs per thread
    v8f accK[2][4] = {};
    v8f accV[2][4] = {};
    h8 aReg[2];
    f4 kReg[2][4], vReg[2][4];

    // prologue: load tile 0
    #pragma unroll
    for (int t = 0; t < 2; ++t)
        aReg[t] = *(const h8*)&A[(size_t)(m0 + ar) * K + (0 + ac + t * 8)];
    load_w_tile(Wk, N, 0, n0, tid, kReg);
    load_w_tile(Wv, N, 0, n0, tid, vReg);

    for (int k0 = 0; k0 < K; k0 += TK) {
        #pragma unroll
        for (int t = 0; t < 2; ++t)
            *(h8*)&As[ar * LDP + ac + t * 8] = aReg[t];
        store_w_tile(Bks, tid, kReg);
        store_w_tile(Bvs, tid, vReg);
        __syncthreads();

        // prefetch next tile while WMMAs run
        if (k0 + TK < K) {
            #pragma unroll
            for (int t = 0; t < 2; ++t)
                aReg[t] = *(const h8*)&A[(size_t)(m0 + ar) * K + (k0 + TK + ac + t * 8)];
            load_w_tile(Wk, N, k0 + TK, n0, tid, kReg);
            load_w_tile(Wv, N, k0 + TK, n0, tid, vReg);
        }

        v16h af[2], bk[4], bv[4];
        #pragma unroll
        for (int i = 0; i < 2; ++i) {
            const _Float16* ap = &As[(wm + i * 16 + mrow) * LDP + half * 16];
            #pragma unroll
            for (int e = 0; e < 8; ++e) { af[i][e] = ap[e]; af[i][e + 8] = ap[8 + e]; }
        }
        #pragma unroll
        for (int j = 0; j < 4; ++j) {
            const _Float16* bkp = &Bks[(wn + j * 16 + mrow) * LDP + half * 16];
            const _Float16* bvp = &Bvs[(wn + j * 16 + mrow) * LDP + half * 16];
            #pragma unroll
            for (int e = 0; e < 16; ++e) { bk[j][e] = bkp[e]; bv[j][e] = bvp[e]; }
        }
        #pragma unroll
        for (int i = 0; i < 2; ++i)
            #pragma unroll
            for (int j = 0; j < 4; ++j) {
                accK[i][j] = __builtin_amdgcn_wmma_f32_16x16x32_f16(
                    false, af[i], false, bk[j], (short)0, accK[i][j], false, false);
                accV[i][j] = __builtin_amdgcn_wmma_f32_16x16x32_f16(
                    false, af[i], false, bv[j], (short)0, accV[i][j], false, false);
            }
        __syncthreads();
    }

    const int ncol = lane & 15, rbase = (lane >> 4) * 8;
    #pragma unroll
    for (int i = 0; i < 2; ++i)
        #pragma unroll
        for (int j = 0; j < 4; ++j)
            #pragma unroll
            for (int v = 0; v < 8; ++v) {
                size_t off = (size_t)(m0 + wm + i * 16 + rbase + v) * N
                           + (n0 + wn + j * 16 + ncol);
                Ko[off] = (_Float16)accK[i][j][v];
                Vo[off] = (_Float16)accV[i][j][v];
            }
}

// ---------------------------------------------------------------------------
// Kernel 4: attention over G=18 connections. One wave32 per (token, head).
// ---------------------------------------------------------------------------
__global__ __launch_bounds__(128)
void attn_kernel(const float* __restrict__ Q, const _Float16* __restrict__ Kw,
                 const _Float16* __restrict__ Vw, float* __restrict__ out) {
    int wid   = blockIdx.x * 4 + (threadIdx.x >> 5);
    int lane  = threadIdx.x & 31;
    int head  = wid & (NH_ - 1);
    int token = wid >> 3;

    const float* q = Q + (size_t)token * H_ + head * D_;
    f2 qv = *(const f2*)&q[2 * lane];

    float scores[G_];
    const _Float16* kbase = Kw + (size_t)token * GK_ + head * D_;
    #pragma unroll
    for (int g = 0; g < G_; ++g) {
        h2 kv = *(const h2*)&kbase[g * H_ + 2 * lane];
        float p = qv[0] * (float)kv[0] + qv[1] * (float)kv[1];
        #pragma unroll
        for (int off = 16; off > 0; off >>= 1) p += __shfl_xor(p, off, 32);
        scores[g] = p * 0.125f;               // 1/sqrt(64)
    }
    float mx = scores[0];
    #pragma unroll
    for (int g = 1; g < G_; ++g) mx = fmaxf(mx, scores[g]);
    float ssum = 0.f;
    #pragma unroll
    for (int g = 0; g < G_; ++g) { scores[g] = __expf(scores[g] - mx); ssum += scores[g]; }
    float inv = 1.0f / ssum;

    float o0 = 0.f, o1 = 0.f;
    const _Float16* vbase = Vw + (size_t)token * GK_ + head * D_;
    #pragma unroll
    for (int g = 0; g < G_; ++g) {
        h2 vv = *(const h2*)&vbase[g * H_ + 2 * lane];
        float a = scores[g] * inv;
        o0 += a * (float)vv[0];
        o1 += a * (float)vv[1];
    }
    f2 ov; ov[0] = o0; ov[1] = o1;
    *(f2*)&out[(size_t)token * H_ + head * D_ + 2 * lane] = ov;
}

// ---------------------------------------------------------------------------
extern "C" void kernel_launch(void* const* d_in, const int* in_sizes, int n_in,
                              void* d_out, int out_size, void* d_ws, size_t ws_size,
                              hipStream_t stream) {
    (void)in_sizes; (void)n_in; (void)out_size; (void)ws_size;
    const float* hidden = (const float*)d_in[0];
    const int*   conn   = (const int*)d_in[1];
    const float* Wq     = (const float*)d_in[2];
    const float* Wk     = (const float*)d_in[3];
    const float* Wv     = (const float*)d_in[4];
    const float* Wo     = (const float*)d_in[5];
    float*       outp   = (float*)d_out;

    // workspace carve-out (~486 MB)
    _Float16* Abuf = (_Float16*)d_ws;                       // 8192*9216 f16
    _Float16* Kbuf = Abuf + (size_t)M_ * GK_;               // 8192*9216 f16
    _Float16* Vbuf = Kbuf + (size_t)M_ * GK_;               // 8192*9216 f16
    float*    Qbuf = (float*)(Vbuf + (size_t)M_ * GK_);     // 8192*512  f32
    float*    AObuf = Qbuf + (size_t)M_ * H_;               // 8192*512  f32

    // 1) gather connected tokens into f16 activation matrix
    gather_kernel<<<B_ * S_ * G_, 128, 0, stream>>>(hidden, conn, Abuf);

    // 2) Q = hidden @ Wq
    gemm_f32_wmma<<<dim3(M_ / TM, H_ / TN), 128, 0, stream>>>(
        hidden, Wq, Qbuf, M_, H_, H_);

    // 3) K,V = A @ {Wk, Wv}   (dominant GEMM pair, weights streamed once)
    gemm_kv_wmma<<<dim3(M_ / TM, GK_ / TN), 128, 0, stream>>>(
        Abuf, Wk, Wv, Kbuf, Vbuf);

    // 4) softmax attention over the G=18 connections
    attn_kernel<<<(M_ * NH_) / 4, 128, 0, stream>>>(Qbuf, Kbuf, Vbuf, AObuf);

    // 5) out = attn_out @ Wo
    gemm_f32_wmma<<<dim3(M_ / TM, H_ / TN), 128, 0, stream>>>(
        AObuf, Wo, outp, M_, H_, H_);
}
